// GRUDecoder_75179107549419
// MI455X (gfx1250) — matrix operand order repaired and checked by reading, not compiled, not run on hardware
//
#include <hip/hip_runtime.h>

// MI455X / gfx1250, wave32. GRU (H=64) over T=1024, B=8192, then FC to 256.
// Recurrent matmul on v_wmma_f32_16x16x32_f16 with register-resident weights.
// Activations on the gfx1250 hardware tanh unit (v_tanh_f32) when available.

typedef __attribute__((ext_vector_type(16))) _Float16 v16h;
typedef __attribute__((ext_vector_type(8)))  _Float16 v8h;
typedef __attribute__((ext_vector_type(8)))  float    v8f;
typedef __attribute__((ext_vector_type(4)))  float    v4f;

#define HID   64
#define TLEN  1024
#define OUTN  256
#define CT    128   // x staging chunk (timesteps)

union AFrag { v16h v; v8h h[2]; };

#if __has_builtin(__builtin_amdgcn_tanhf)
#define TANHF(x) __builtin_amdgcn_tanhf(x)
#elif __has_builtin(__builtin_amdgcn_tanh_f32)
#define TANHF(x) __builtin_amdgcn_tanh_f32(x)
#else
// fallback: tanh(x) = 2/(1+2^(-2x*log2 e)) - 1
#define TANHF(x) (2.0f * __builtin_amdgcn_rcpf(1.0f + \
                  __builtin_amdgcn_exp2f((x) * -2.88539008178f)) - 1.0f)
#endif

__device__ __forceinline__ void loadA(const _Float16* hw, int N, int hi,
                                      AFrag& a0, AFrag& a1) {
  // A fragment (16x32 f16): lane holds row M=N; elem e -> K=16*hi+e.
  // Chunk c adds c*32 to K.
  a0.h[0] = *(const v8h*)&hw[N*64 +  0 + hi*16];
  a0.h[1] = *(const v8h*)&hw[N*64 +  8 + hi*16];
  a1.h[0] = *(const v8h*)&hw[N*64 + 32 + hi*16];
  a1.h[1] = *(const v8h*)&hw[N*64 + 40 + hi*16];
}

__global__ __launch_bounds__(128) void gru_decoder_kernel(
    const float* __restrict__ x,    const float* __restrict__ w_ih,
    const float* __restrict__ w_hh, const float* __restrict__ b_ih,
    const float* __restrict__ b_hh, const float* __restrict__ fc_w,
    const float* __restrict__ fc_b, float* __restrict__ out)
{
  // fragments: 24 GRU (12 col-tiles x 2 K-chunks) + 32 FC (16 x 2), 512 f16 each
  __shared__ _Float16 fragS[56 * 512];      // 56 KB
  __shared__ _Float16 hbufS[4 * 16 * 64];   // per-wave 16x64 f16 h tile, 8 KB
  __shared__ float    xbufS[4 * CT * 16];   // per-wave x staging [t][row], 32 KB

  const int tid  = threadIdx.x;
  const int wave = tid >> 5;
  const int lane = tid & 31;
  const int hi   = lane >> 4;   // lane half
  const int N    = lane & 15;   // col within tile / row for A
  const int rowBase = blockIdx.x * 64 + wave * 16;

  // ---- stage weight fragments in pre-swizzled B layout ----
  // B fragment (32x16 f16): lane l, elem e -> K = e + 16*(l>>4), col = l&15
  for (int idx = tid; idx < 56 * 512; idx += 128) {
    const int f = idx >> 9;
    const int r = idx & 511;
    const int l = r >> 4;
    const int e = r & 15;
    const int K = e + ((l >> 4) << 4);
    const int n = l & 15;
    float val;
    if (f < 24) {
      const int j = f >> 1, c = f & 1;               // gate col-tile, K-chunk
      val = w_hh[(j * 16 + n) * HID + c * 32 + K];   // B[k][g] = w_hh[g][k]
    } else {
      const int ff = f - 24;
      const int j = ff >> 1, c = ff & 1;             // out col-tile, K-chunk
      val = fc_w[(j * 16 + n) * HID + c * 32 + K];
    }
    fragS[idx] = (_Float16)val;
  }
  for (int idx = tid; idx < 4 * 1024; idx += 128) hbufS[idx] = (_Float16)0.0f;
  __syncthreads();

  _Float16* hw = &hbufS[wave * 1024];
  float*    xw = &xbufS[wave * CT * 16];

  // ---- register-resident recurrent weights: 24 fragments = 192 VGPRs ----
  v16h bfr[24];
  #pragma unroll
  for (int i = 0; i < 24; ++i)
    bfr[i] = *(const v16h*)&fragS[i * 512 + lane * 16];

  // ---- per-lane loop-invariant gate constants ----
  // r,z gates use half-scaled weights/biases so sigmoid(a)=0.5*tanh(a/2)+0.5
  // needs only one extra fma per activation.
  float whalf[8], bhalf[8], wihn[4], bihn[4], bhhn[4];
  #pragma unroll
  for (int j = 0; j < 12; ++j) {
    const int col = j * 16 + N;
    if (j < 8) {
      whalf[j] = 0.5f * w_ih[col];
      bhalf[j] = 0.5f * (b_ih[col] + b_hh[col]);
    } else {
      wihn[j - 8] = w_ih[col];
      bihn[j - 8] = b_ih[col];
      bhhn[j - 8] = b_hh[col];
    }
  }

  // h state in f32, C layout: hst[j][e] = h[row e+8*hi][col j*16+N]
  v8f hst[4];
  #pragma unroll
  for (int j = 0; j < 4; ++j) hst[j] = (v8f){0,0,0,0,0,0,0,0};

  #pragma unroll 1
  for (int t = 0; t < TLEN; ++t) {
    const int tt = t & (CT - 1);
    if (tt == 0) {
      // refill per-wave x staging: coalesced global read, transposed LDS write
      #pragma unroll 4
      for (int i = 0; i < 64; ++i) {
        const int idx = i * 32 + lane;
        const int row = idx >> 7;
        const int c2  = idx & (CT - 1);
        xw[c2 * 16 + row] = x[(rowBase + row) * TLEN + t + c2];
      }
    }

    // broadcast x for this lane's 8 rows
    v4f xa = *(const v4f*)&xw[tt * 16 + hi * 8];
    v4f xb = *(const v4f*)&xw[tt * 16 + hi * 8 + 4];
    float xv[8];
    #pragma unroll
    for (int e = 0; e < 4; ++e) { xv[e] = xa[e]; xv[e + 4] = xb[e]; }

    // A fragments from f16 h tile (cross-lane transpose via LDS)
    AFrag a0, a1;
    loadA(hw, N, hi, a0, a1);

    // gh = h @ w_hh^T : 24 WMMAs
    v8f g[12];
    #pragma unroll
    for (int j = 0; j < 12; ++j) {
      v8f z = (v8f){0,0,0,0,0,0,0,0};
      z = __builtin_amdgcn_wmma_f32_16x16x32_f16(false, a0.v, false, bfr[2*j],
                                                 (short)0, z, false, false);
      z = __builtin_amdgcn_wmma_f32_16x16x32_f16(false, a1.v, false, bfr[2*j+1],
                                                 (short)0, z, false, false);
      g[j] = z;
    }

    // gates (PyTorch order r,z,n): h = (1-z)*n + z*h, n = tanh(xn + r*hn)
    // sigmoid(a) = 0.5*tanh(0.5*a) + 0.5, halves prefolded into whalf/bhalf
    #pragma unroll
    for (int j = 0; j < 4; ++j) {
      #pragma unroll
      for (int e = 0; e < 8; ++e) {
        const float ur = 0.5f * g[j][e]     + (xv[e] * whalf[j]     + bhalf[j]);
        const float rr = TANHF(ur) * 0.5f + 0.5f;
        const float uz = 0.5f * g[j + 4][e] + (xv[e] * whalf[j + 4] + bhalf[j + 4]);
        const float zz = TANHF(uz) * 0.5f + 0.5f;
        const float hn = g[j + 8][e] + bhhn[j];
        const float xn = xv[e] * wihn[j] + bihn[j];
        const float nn = TANHF(rr * hn + xn);
        const float hv = zz * (hst[j][e] - nn) + nn;
        hst[j][e] = hv;
        hw[(e + hi * 8) * 64 + j * 16 + N] = (_Float16)hv;  // f16 copy for next A
      }
    }
  }

  // ---- FC: out = hT @ fc_w^T + fc_b, 32 WMMAs ----
  AFrag a0, a1;
  loadA(hw, N, hi, a0, a1);
  #pragma unroll 1
  for (int j = 0; j < 16; ++j) {
    v16h f0 = *(const v16h*)&fragS[(24 + 2*j)     * 512 + lane * 16];
    v16h f1 = *(const v16h*)&fragS[(24 + 2*j + 1) * 512 + lane * 16];
    v8f o = (v8f){0,0,0,0,0,0,0,0};
    o = __builtin_amdgcn_wmma_f32_16x16x32_f16(false, a0.v, false, f0,
                                               (short)0, o, false, false);
    o = __builtin_amdgcn_wmma_f32_16x16x32_f16(false, a1.v, false, f1,
                                               (short)0, o, false, false);
    const float fb = fc_b[j * 16 + N];
    #pragma unroll
    for (int e = 0; e < 8; ++e)
      out[(rowBase + e + hi * 8) * OUTN + j * 16 + N] = o[e] + fb;
  }
}

extern "C" void kernel_launch(void* const* d_in, const int* in_sizes, int n_in,
                              void* d_out, int out_size, void* d_ws, size_t ws_size,
                              hipStream_t stream) {
  const float* x    = (const float*)d_in[0];
  const float* w_ih = (const float*)d_in[1];
  const float* w_hh = (const float*)d_in[2];
  const float* b_ih = (const float*)d_in[3];
  const float* b_hh = (const float*)d_in[4];
  const float* fc_w = (const float*)d_in[5];
  const float* fc_b = (const float*)d_in[6];
  float* out = (float*)d_out;

  dim3 grid(8192 / 64);   // 64 batch rows per block (4 waves x 16 rows)
  dim3 block(128);
  gru_decoder_kernel<<<grid, block, 0, stream>>>(x, w_ih, w_hh, b_ih, b_hh,
                                                 fc_w, fc_b, out);
}
